// DiTAttention_43044162241205
// MI455X (gfx1250) — compile-verified, matrix operand order checked by reading, not run on hardware
//
#include <hip/hip_runtime.h>
#include <hip/hip_fp16.h>
#include <math.h>

typedef __attribute__((ext_vector_type(16))) _Float16 v16h;
typedef __attribute__((ext_vector_type(8)))  float    v8f;

#define C_S   768
#define C_Z   128
#define DH    32
#define NH    24
#define SQ    1024
#define BB    2
#define EPS   1e-5f
#define NEG_BIG (-30000.0f)

// workspace layout (bytes). total required: 72 MB
#define EMB_OFF   0u            // f32 [2][2304]              (18 KB)
#define WQT_OFF   0x00010000u   // f16 [768][768] transposed  (1.125 MB)
#define WKT_OFF   0x00140000u
#define WVT_OFF   0x00280000u
#define WOT_OFF   0x003C0000u
#define BSN_OFF   0x00500000u   // f16 [2048][768]            (3 MB)
#define Q_OFF     (8u<<20)      // f16 [B][H][S][D]           (3 MB)
#define K_OFF     (12u<<20)     // f16 [B][H][S][D]
#define VT_OFF    (16u<<20)     // f16 [B][H][D][S]  (transposed for PV WMMA)
#define AO_OFF    (20u<<20)     // f16 [2048][768]            (3 MB)
#define BIAS_OFF  (24u<<20)     // f16 [H][S][S]              (48 MB)

union HU  { v16h h; unsigned u[8]; };

// ---------------------------------------------------------------------------
// K0: transpose + f32->f16 convert all weight matrices to [n][k] layout
// grid (24, 24, 4) x 256
// ---------------------------------------------------------------------------
__global__ void k_wprep(const float* __restrict__ wq, const float* __restrict__ wk,
                        const float* __restrict__ wv, const float* __restrict__ wo,
                        _Float16* __restrict__ wqt, _Float16* __restrict__ wkt,
                        _Float16* __restrict__ wvt, _Float16* __restrict__ wot) {
  __shared__ float tile[32][33];
  int mz = blockIdx.z;
  const float* W  = (mz == 0) ? wq  : (mz == 1) ? wk  : (mz == 2) ? wv  : wo;
  _Float16*    WT = (mz == 0) ? wqt : (mz == 1) ? wkt : (mz == 2) ? wvt : wot;
  int k0 = blockIdx.x * 32, n0 = blockIdx.y * 32;
  int tc = threadIdx.x & 31, tr = threadIdx.x >> 5;   // 32 x 8
#pragma unroll
  for (int i = 0; i < 4; ++i)
    tile[tr + i * 8][tc] = W[(long long)(k0 + tr + i * 8) * C_S + n0 + tc];
  __syncthreads();
#pragma unroll
  for (int i = 0; i < 4; ++i)
    WT[(long long)(n0 + tr + i * 8) * C_S + k0 + tc] = (_Float16)tile[tc][tr + i * 8];
}

// ---------------------------------------------------------------------------
// K1: emb = silu(t) @ w_adaln + b_adaln        grid (9, 2) x 256
// ---------------------------------------------------------------------------
__global__ void k_adaln(const float* __restrict__ t, const float* __restrict__ w,
                        const float* __restrict__ b, float* __restrict__ emb) {
  int j = blockIdx.x * 256 + threadIdx.x;   // 0..2303
  int bb = blockIdx.y;
  float acc = b[j];
  for (int c = 0; c < C_S; ++c) {
    float tv = t[bb * C_S + c];
    float s  = tv / (1.0f + __expf(-tv));
    acc = fmaf(s, w[c * 3 * C_S + j], acc);
  }
  emb[bb * 3 * C_S + j] = acc;
}

// ---------------------------------------------------------------------------
// K2: bs_norm = LN(bs)*(1+scale)+shift -> f16   grid (1024, 2) x 256
// ---------------------------------------------------------------------------
__global__ void k_bsnorm(const float* __restrict__ bs, const float* __restrict__ emb,
                         _Float16* __restrict__ bsn) {
  __shared__ float red[256];
  int row = blockIdx.x, bb = blockIdx.y, tid = threadIdx.x;
  const float* x = bs + (long long)(bb * SQ + row) * C_S;
  float v0 = x[tid], v1 = x[tid + 256], v2 = x[tid + 512];
  red[tid] = v0 + v1 + v2;
  __syncthreads();
  for (int o = 128; o > 0; o >>= 1) { if (tid < o) red[tid] += red[tid + o]; __syncthreads(); }
  float mu = red[0] * (1.0f / C_S);
  __syncthreads();
  red[tid] = v0 * v0 + v1 * v1 + v2 * v2;
  __syncthreads();
  for (int o = 128; o > 0; o >>= 1) { if (tid < o) red[tid] += red[tid + o]; __syncthreads(); }
  float var = red[0] * (1.0f / C_S) - mu * mu;
  float rs  = rsqrtf(var + EPS);
  const float* sh = emb + bb * 3 * C_S;
  const float* sc = sh + C_S;
  _Float16* op = bsn + (long long)(bb * SQ + row) * C_S;
  for (int c = tid; c < C_S; c += 256) {
    float y = (x[c] - mu) * rs;
    op[c] = (_Float16)(y * (1.0f + sc[c]) + sh[c]);
  }
}

// ---------------------------------------------------------------------------
// K3: bias[h,i,j] = (LN(z[i,j,:])*lw+lb) . w_z[:,h] + mask   (single z pass)
// grid (4096) x 256, one thread per (i,j)
// ---------------------------------------------------------------------------
__global__ void k_zbias(const float* __restrict__ z, const int* __restrict__ zmask,
                        const float* __restrict__ lw, const float* __restrict__ lb,
                        const float* __restrict__ wz, _Float16* __restrict__ biasout) {
  __shared__ float wwz[C_Z * NH];
  __shared__ float swz[NH], bwz[NH];
  int tid = threadIdx.x;
  for (int idx = tid; idx < C_Z * NH; idx += 256) {
    int c = idx / NH, h = idx % NH;
    wwz[idx] = lw[c] * wz[c * NH + h];
  }
  if (tid < NH) {
    float s = 0.f, bsum = 0.f;
    for (int c = 0; c < C_Z; ++c) { float w = wz[c * NH + tid]; s += lw[c] * w; bsum += lb[c] * w; }
    swz[tid] = s; bwz[tid] = bsum;
  }
  __syncthreads();
  long long pid = (long long)blockIdx.x * 256 + tid;
  int i = (int)(pid >> 10), j = (int)(pid & 1023);
  const float* x = z + ((long long)i * SQ + j) * C_Z;
  __builtin_prefetch(x, 0, 0);
  float acc[NH];
#pragma unroll
  for (int h = 0; h < NH; ++h) acc[h] = 0.f;
  float s = 0.f, ss = 0.f;
  for (int c = 0; c < C_Z; ++c) {
    float xv = x[c];
    s += xv; ss += xv * xv;
#pragma unroll
    for (int h = 0; h < NH; ++h) acc[h] = fmaf(xv, wwz[c * NH + h], acc[h]);
  }
  float mu  = s * (1.0f / C_Z);
  float var = ss * (1.0f / C_Z) - mu * mu;
  float rs  = rsqrtf(var + EPS);
  float mb  = (zmask[(long long)i * SQ + j] > 0) ? 0.0f : NEG_BIG;
#pragma unroll
  for (int h = 0; h < NH; ++h) {
    float bv = rs * (acc[h] - mu * swz[h]) + bwz[h] + mb;
    biasout[(long long)h * SQ * SQ + (long long)i * SQ + j] = (_Float16)bv;
  }
}

// ---------------------------------------------------------------------------
// K4: [q|k|v] = bs_norm @ w via WMMA (f16 transposed weights).
// q,k -> (B,H,S,D);  v -> (B,H,D,S) transposed for the PV WMMA B-fragment.
// grid (32, 144) x 128 (4 waves, one 16x16 tile each)
// ---------------------------------------------------------------------------
__global__ void k_qkv(const _Float16* __restrict__ bsn, const _Float16* __restrict__ wqt,
                      const _Float16* __restrict__ wkt, const _Float16* __restrict__ wvt,
                      _Float16* __restrict__ qo, _Float16* __restrict__ ko,
                      _Float16* __restrict__ vt) {
  int lane = threadIdx.x & 31;
  int wv_id = threadIdx.x >> 5;
  int mtile = blockIdx.x * 4 + wv_id;     // 0..127
  int ntile = blockIdx.y;                 // 0..143
  int ln = lane & 15, hf = lane >> 4;
  int nglob = ntile * 16 + ln;
  int proj  = nglob / C_S;                // uniform per tile
  int ncol  = nglob % C_S;
  const _Float16* WT = (proj == 0) ? wqt : ((proj == 1) ? wkt : wvt);
  int mrow = mtile * 16 + ln;
  v8f acc = {0.f, 0.f, 0.f, 0.f, 0.f, 0.f, 0.f, 0.f};
  for (int kk = 0; kk < C_S; kk += 32) {
    HU a, bf;
#pragma unroll
    for (int v = 0; v < 8; ++v) {
      int kidx = (v & 3) * 2 + (v >> 2) * 16 + hf * 8;
      a.u[v]  = *(const unsigned*)(bsn + (long long)mrow * C_S + kk + kidx);
      bf.u[v] = *(const unsigned*)(WT  + (long long)ncol * C_S + kk + kidx);
    }
    acc = __builtin_amdgcn_wmma_f32_16x16x32_f16(false, a.h, false, bf.h,
                                                 (short)0, acc, false, false);
  }
  int hh = ncol >> 5, d = ncol & 31;
  if (proj == 2) {
#pragma unroll
    for (int r = 0; r < 8; ++r) {
      int m = mtile * 16 + r + hf * 8;
      int bb = m >> 10, sidx = m & 1023;
      vt[((long long)(bb * NH + hh) * DH + d) * SQ + sidx] = (_Float16)acc[r];
    }
  } else {
    _Float16* dst = (proj == 0) ? qo : ko;
#pragma unroll
    for (int r = 0; r < 8; ++r) {
      int m = mtile * 16 + r + hf * 8;
      int bb = m >> 10, sidx = m & 1023;
      dst[(((long long)(bb * NH + hh) * SQ + sidx) << 5) + d] = (_Float16)acc[r];
    }
  }
}

// ---------------------------------------------------------------------------
// K5: RMS norm of q and k over D=32 (wave32 shuffle)  grid (6144, 2) x 256
// ---------------------------------------------------------------------------
__global__ void k_rms(_Float16* __restrict__ q, _Float16* __restrict__ k,
                      const float* __restrict__ wqr, const float* __restrict__ wkr) {
  int tid  = threadIdx.x;
  int lane = tid & 31;
  long long row = (long long)blockIdx.x * 8 + (tid >> 5);
  _Float16* p = (blockIdx.y == 0) ? q : k;
  const float* w = (blockIdx.y == 0) ? wqr : wkr;
  float x = (float)p[row * DH + lane];
  float ss = x * x;
  for (int o = 16; o > 0; o >>= 1) ss += __shfl_xor(ss, o, 32);
  float rs = rsqrtf(ss * (1.0f / DH) + EPS);
  p[row * DH + lane] = (_Float16)(x * rs * w[lane]);
}

// ---------------------------------------------------------------------------
// K6: flash attention.  grid (8, 24, 2) x 256 : 8 waves x 16-query tiles
// ---------------------------------------------------------------------------
__global__ void __launch_bounds__(256)
k_attn(const _Float16* __restrict__ q, const _Float16* __restrict__ k,
       const _Float16* __restrict__ vt, const _Float16* __restrict__ bias,
       const float* __restrict__ beta, _Float16* __restrict__ ao) {
  __shared__ _Float16 pstage[8][16 * 32];
  int tid = threadIdx.x;
  int wv_id = tid >> 5, lane = tid & 31, ln = lane & 15, hf = lane >> 4;
  int hh = blockIdx.y, bb = blockIdx.z;
  int q0 = blockIdx.x * 128 + wv_id * 16;
  long long headbase = (long long)(bb * NH + hh) * SQ * DH;  // same extent for vt

  HU qf;
#pragma unroll
  for (int vv = 0; vv < 8; ++vv) {
    int kidx = (vv & 3) * 2 + (vv >> 2) * 16 + hf * 8;
    qf.u[vv] = *(const unsigned*)(q + headbase + (long long)(q0 + ln) * DH + kidx);
  }
  float mi[8], li[8];
  v8f o0 = {0.f,0.f,0.f,0.f,0.f,0.f,0.f,0.f};
  v8f o1 = {0.f,0.f,0.f,0.f,0.f,0.f,0.f,0.f};
#pragma unroll
  for (int r = 0; r < 8; ++r) { mi[r] = -1e30f; li[r] = 0.f; }
  const _Float16* biasb = bias + (long long)hh * SQ * SQ;
  const float*    betab = beta + (long long)bb * SQ * SQ;
  const float scl = 0.17677669529663687f;   // 1/sqrt(32)

  for (int kt = 0; kt < SQ; kt += 32) {
    // K^T fragments: lane = key, regs sweep d; (b,h,s,d) layout is contiguous in d
    HU kf0, kf1;
#pragma unroll
    for (int vv = 0; vv < 8; ++vv) {
      int didx = (vv & 3) * 2 + (vv >> 2) * 16 + hf * 8;
      kf0.u[vv] = *(const unsigned*)(k + headbase + (long long)(kt + ln) * DH + didx);
      kf1.u[vv] = *(const unsigned*)(k + headbase + (long long)(kt + 16 + ln) * DH + didx);
    }
    v8f z8 = {0.f,0.f,0.f,0.f,0.f,0.f,0.f,0.f};
    v8f s0 = __builtin_amdgcn_wmma_f32_16x16x32_f16(false, qf.h, false, kf0.h, (short)0, z8, false, false);
    v8f s1 = __builtin_amdgcn_wmma_f32_16x16x32_f16(false, qf.h, false, kf1.h, (short)0, z8, false, false);
#pragma unroll
    for (int r = 0; r < 8; ++r) {
      int row = q0 + r + hf * 8;
      long long base = (long long)row * SQ + kt + ln;
      float b0 = (float)biasb[base]      + betab[base];
      float b1 = (float)biasb[base + 16] + betab[base + 16];
      s0[r] = fmaf(s0[r], scl, b0);
      s1[r] = fmaf(s1[r], scl, b1);
    }
#pragma unroll
    for (int r = 0; r < 8; ++r) {
      float mx = fmaxf(s0[r], s1[r]);
      for (int o = 8; o > 0; o >>= 1) mx = fmaxf(mx, __shfl_xor(mx, o, 32));
      float mnew = fmaxf(mi[r], mx);
      float p0 = __expf(s0[r] - mnew), p1 = __expf(s1[r] - mnew);
      float rsum = p0 + p1;
      for (int o = 8; o > 0; o >>= 1) rsum += __shfl_xor(rsum, o, 32);
      float alpha = __expf(mi[r] - mnew);
      li[r] = li[r] * alpha + rsum;
      o0[r] *= alpha; o1[r] *= alpha;
      mi[r] = mnew;
      s0[r] = p0; s1[r] = p1;
    }
    // restage P (C-frag layout) -> A-frag layout through LDS
    __syncthreads();
#pragma unroll
    for (int r = 0; r < 8; ++r) {
      int qloc = r + hf * 8;
      pstage[wv_id][qloc * 32 + ln]      = (_Float16)s0[r];
      pstage[wv_id][qloc * 32 + 16 + ln] = (_Float16)s1[r];
    }
    __syncthreads();
    HU pf;
#pragma unroll
    for (int vv = 0; vv < 8; ++vv) {
      int kidx = (vv & 3) * 2 + (vv >> 2) * 16 + hf * 8;
      pf.u[vv] = *(const unsigned*)&pstage[wv_id][ln * 32 + kidx];
    }
    // V fragments from transposed (b,h,d,s) layout: lane = d, regs sweep key
    HU vf0, vf1;
#pragma unroll
    for (int vv = 0; vv < 8; ++vv) {
      int kidx = (vv & 3) * 2 + (vv >> 2) * 16 + hf * 8;
      vf0.u[vv] = *(const unsigned*)(vt + headbase + (long long)ln * SQ + kt + kidx);
      vf1.u[vv] = *(const unsigned*)(vt + headbase + (long long)(16 + ln) * SQ + kt + kidx);
    }
    o0 = __builtin_amdgcn_wmma_f32_16x16x32_f16(false, pf.h, false, vf0.h, (short)0, o0, false, false);
    o1 = __builtin_amdgcn_wmma_f32_16x16x32_f16(false, pf.h, false, vf1.h, (short)0, o1, false, false);
  }
#pragma unroll
  for (int r = 0; r < 8; ++r) {
    int row = q0 + r + hf * 8;
    float inv = 1.0f / li[r];
    _Float16* dst = ao + (long long)(bb * SQ + row) * C_S + hh * DH;
    dst[ln]      = (_Float16)(o0[r] * inv);
    dst[16 + ln] = (_Float16)(o1[r] * inv);
  }
}

// ---------------------------------------------------------------------------
// K7: out = (attn_out @ w_o + b_o) * gate  -> f32 d_out  grid (32,48) x 128
// ---------------------------------------------------------------------------
__global__ void k_oproj(const _Float16* __restrict__ ao, const _Float16* __restrict__ wot,
                        const float* __restrict__ bo, const float* __restrict__ emb,
                        float* __restrict__ out) {
  int lane = threadIdx.x & 31;
  int wv_id = threadIdx.x >> 5;
  int mtile = blockIdx.x * 4 + wv_id;
  int ntile = blockIdx.y;                // 0..47
  int ln = lane & 15, hf = lane >> 4;
  int ncol = ntile * 16 + ln;
  int mrow = mtile * 16 + ln;
  v8f acc = {0.f,0.f,0.f,0.f,0.f,0.f,0.f,0.f};
  for (int kk = 0; kk < C_S; kk += 32) {
    HU a, bf;
#pragma unroll
    for (int v = 0; v < 8; ++v) {
      int kidx = (v & 3) * 2 + (v >> 2) * 16 + hf * 8;
      a.u[v]  = *(const unsigned*)(ao  + (long long)mrow * C_S + kk + kidx);
      bf.u[v] = *(const unsigned*)(wot + (long long)ncol * C_S + kk + kidx);
    }
    acc = __builtin_amdgcn_wmma_f32_16x16x32_f16(false, a.h, false, bf.h,
                                                 (short)0, acc, false, false);
  }
  float bv = bo[ncol];
#pragma unroll
  for (int r = 0; r < 8; ++r) {
    int m = mtile * 16 + r + hf * 8;
    int bb = m >> 10;
    float g = emb[bb * 3 * C_S + 2 * C_S + ncol];
    out[(long long)m * C_S + ncol] = (acc[r] + bv) * g;
  }
}

// ---------------------------------------------------------------------------
extern "C" void kernel_launch(void* const* d_in, const int* in_sizes, int n_in,
                              void* d_out, int out_size, void* d_ws, size_t ws_size,
                              hipStream_t stream) {
  const float* bs      = (const float*)d_in[0];
  const float* z       = (const float*)d_in[1];
  const float* t       = (const float*)d_in[2];
  const float* beta    = (const float*)d_in[3];
  const int*   zmask   = (const int*)d_in[4];
  const float* w_adaln = (const float*)d_in[5];
  const float* b_adaln = (const float*)d_in[6];
  const float* ln_z_w  = (const float*)d_in[7];
  const float* ln_z_b  = (const float*)d_in[8];
  const float* w_q     = (const float*)d_in[9];
  const float* w_k     = (const float*)d_in[10];
  const float* w_v     = (const float*)d_in[11];
  const float* w_z     = (const float*)d_in[12];
  const float* rms_q_w = (const float*)d_in[13];
  const float* rms_k_w = (const float*)d_in[14];
  const float* w_o     = (const float*)d_in[15];
  const float* b_o     = (const float*)d_in[16];
  float* out = (float*)d_out;

  char* ws = (char*)d_ws;
  float*     emb  = (float*)(ws + EMB_OFF);
  _Float16*  wqt  = (_Float16*)(ws + WQT_OFF);
  _Float16*  wkt  = (_Float16*)(ws + WKT_OFF);
  _Float16*  wvt  = (_Float16*)(ws + WVT_OFF);
  _Float16*  wot  = (_Float16*)(ws + WOT_OFF);
  _Float16*  bsn  = (_Float16*)(ws + BSN_OFF);
  _Float16*  qb   = (_Float16*)(ws + Q_OFF);
  _Float16*  kb   = (_Float16*)(ws + K_OFF);
  _Float16*  vtb  = (_Float16*)(ws + VT_OFF);
  _Float16*  ao   = (_Float16*)(ws + AO_OFF);
  _Float16*  bias = (_Float16*)(ws + BIAS_OFF);

  k_wprep<<<dim3(24, 24, 4), 256, 0, stream>>>(w_q, w_k, w_v, w_o, wqt, wkt, wvt, wot);
  k_adaln<<<dim3(9, 2), 256, 0, stream>>>(t, w_adaln, b_adaln, emb);
  k_bsnorm<<<dim3(SQ, BB), 256, 0, stream>>>(bs, emb, bsn);
  k_zbias<<<dim3((SQ * SQ) / 256), 256, 0, stream>>>(z, zmask, ln_z_w, ln_z_b, w_z, bias);
  k_qkv<<<dim3(32, 144), 128, 0, stream>>>(bsn, wqt, wkt, wvt, qb, kb, vtb);
  k_rms<<<dim3((BB * NH * SQ) / 8, 2), 256, 0, stream>>>(qb, kb, rms_q_w, rms_k_w);
  k_attn<<<dim3(SQ / 128, NH, BB), 256, 0, stream>>>(qb, kb, vtb, bias, beta, ao);
  k_oproj<<<dim3(32, 48), 128, 0, stream>>>(ao, wot, b_o, emb, out);
}
